// GATLinkPrediction_88914412962590
// MI455X (gfx1250) — matrix operand order repaired and checked by reading, not compiled, not run on hardware
//
#include <hip/hip_runtime.h>
#include <hip/hip_bf16.h>
#include <math.h>

#define NNODES 50000
#define NEDGES 1600000
#define ETOT   (NEDGES + NNODES)   // self loops appended
#define FDIM   128                 // every layer is 128 -> 128 (2 heads x 64)

typedef __attribute__((ext_vector_type(16))) _Float16 v16h;
typedef __attribute__((ext_vector_type(8)))  _Float16 v8h;
typedef __attribute__((ext_vector_type(8)))  float    v8f;

__device__ __forceinline__ unsigned f2ord(float f) {
  unsigned u = __float_as_uint(f);
  return (u & 0x80000000u) ? ~u : (u | 0x80000000u);
}
__device__ __forceinline__ float ord2f(unsigned u) {
  return (u & 0x80000000u) ? __uint_as_float(u & 0x7FFFFFFFu) : __uint_as_float(~u);
}
__device__ __forceinline__ float lrelu(float x) { return x >= 0.f ? x : 0.2f * x; }

// ---------------------------------------------------------------------------
// One-shot f32 -> f16 feature convert (layer-0 input).
// ---------------------------------------------------------------------------
__global__ void gat_cvt_f16(const float* __restrict__ in, _Float16* __restrict__ out, int n) {
  const int i = blockIdx.x * blockDim.x + threadIdx.x;
  if (i < n) out[i] = (_Float16)in[i];
}

// ---------------------------------------------------------------------------
// Swizzle W (128x128 f32, row-major) into per-lane WMMA B fragments, f16, contiguous k (same order as A):
// Wswz[((kc*8 + t)*32 + lane)*16 + i]  <-  W[(kc*32 + (lane>>4)*16 + i)*128 + t*16 + (lane&15)]
// so each lane's B fragment is one contiguous 32-byte load.
// ---------------------------------------------------------------------------
__global__ void gat_swizzle_w(const float* __restrict__ W, _Float16* __restrict__ Wswz) {
  const int idx = blockIdx.x * blockDim.x + threadIdx.x;   // 32 frags * 32 lanes * 16 = 16384
  if (idx >= 32 * 32 * 16) return;
  const int i    = idx & 15;
  const int lane = (idx >> 4) & 31;
  const int frag = idx >> 9;            // kc*8 + t
  const int t = frag & 7, kc = frag >> 3;
  const int n = t * 16 + (lane & 15);
  const int k = kc * 32 + (lane >> 4) * 16 + i;
  Wswz[idx] = (_Float16)W[k * FDIM + n];
}

// ---------------------------------------------------------------------------
// S = Xf(50000x128 f16) @ Wswz(pre-swizzled f16) via v_wmma_f32_16x16x32_f16.
// 64 threads = 2 waves per block; wave w owns rows [16*bx,16*bx+16) and
// columns [64w, 64w+64) as four 16x16 accumulators, K swept in 4 steps of 32.
// All fragment loads are contiguous 128-bit vector loads.
// ---------------------------------------------------------------------------
__global__ __launch_bounds__(64) void gat_gemm_wmma(const _Float16* __restrict__ Xf,
                                                    const _Float16* __restrict__ Wswz,
                                                    float* __restrict__ S) {
  const int lane = threadIdx.x & 31;
  const int wv   = threadIdx.x >> 5;
  const int row0 = blockIdx.x << 4;
  const int col0 = wv << 6;
  const int m    = lane & 15;
  const int half = lane >> 4;
  const v16h* __restrict__ bfr = (const v16h*)Wswz;

  v8f acc[4] = {};
#pragma unroll
  for (int kc = 0; kc < 4; ++kc) {
    // A fragment: lane m holds row (row0+m); element i -> K = kc*32 + half*16 + i:
    // two contiguous 16B chunks.
    const _Float16* ap = Xf + (row0 + m) * FDIM + kc * 32 + half * 16;
    const v8h alo = *(const v8h*)ap;
    const v8h ahi = *(const v8h*)(ap + 8);
    const v16h a = __builtin_shufflevector(alo, ahi, 0, 1, 2, 3, 4, 5, 6, 7,
                                           8, 9, 10, 11, 12, 13, 14, 15);
#pragma unroll
    for (int ct = 0; ct < 4; ++ct) {
      const v16h b = bfr[(kc * 8 + wv * 4 + ct) * 32 + lane];   // one 32B load
      acc[ct] = __builtin_amdgcn_wmma_f32_16x16x32_f16(
          false, a, false, b, (short)0, acc[ct], false, false);
    }
  }
  // C/D layout: VGPR r holds M = r + 8*half, N = lane%16
#pragma unroll
  for (int ct = 0; ct < 4; ++ct)
#pragma unroll
    for (int r = 0; r < 8; ++r)
      S[(row0 + r + 8 * half) * FDIM + col0 + ct * 16 + m] = acc[ct][r];
}

// ---------------------------------------------------------------------------
// Zero the aggregation buffer; reset segment-max (ordered -inf) and denom.
// ---------------------------------------------------------------------------
__global__ void gat_init(float* __restrict__ agg, unsigned* __restrict__ menc,
                         float* __restrict__ denom) {
  const int i = blockIdx.x * blockDim.x + threadIdx.x;
  if (i < NNODES * FDIM) agg[i] = 0.f;
  if (i < NNODES * 2) { menc[i] = 0x007FFFFFu; /* f2ord(-inf) */ denom[i] = 0.f; }
}

// ---------------------------------------------------------------------------
// alpha_src/alpha_dst per node (wave32 per node; shfl_xor tree reduce).
// a_src/a_dst are [2,64] flat: head h channel c at h*64+c.
// ---------------------------------------------------------------------------
__global__ __launch_bounds__(256) void gat_alpha(const float* __restrict__ S,
                                                 const float* __restrict__ a_src,
                                                 const float* __restrict__ a_dst,
                                                 float* __restrict__ asrc,
                                                 float* __restrict__ adst) {
  const int node = blockIdx.x * 8 + (threadIdx.x >> 5);
  if (node >= NNODES) return;
  const int l = threadIdx.x & 31;
  const float* srow = S + node * FDIM;
  const float s0 = srow[l], s1 = srow[l + 32], s2 = srow[l + 64], s3 = srow[l + 96];
  float ps0 = s0 * a_src[l]      + s1 * a_src[l + 32];
  float ps1 = s2 * a_src[64 + l] + s3 * a_src[96 + l];
  float pd0 = s0 * a_dst[l]      + s1 * a_dst[l + 32];
  float pd1 = s2 * a_dst[64 + l] + s3 * a_dst[96 + l];
#pragma unroll
  for (int off = 16; off; off >>= 1) {
    ps0 += __shfl_xor(ps0, off, 32);
    ps1 += __shfl_xor(ps1, off, 32);
    pd0 += __shfl_xor(pd0, off, 32);
    pd1 += __shfl_xor(pd1, off, 32);
  }
  if (l == 0) {
    asrc[node * 2] = ps0; asrc[node * 2 + 1] = ps1;
    adst[node * 2] = pd0; adst[node * 2 + 1] = pd1;
  }
}

__device__ __forceinline__ void edge_sd(const int* __restrict__ ei, int e, int& s, int& d) {
  if (e < NEDGES) { s = ei[e]; d = ei[NEDGES + e]; }
  else            { s = d = e - NEDGES; }
}

// ---------------------------------------------------------------------------
// Segment max of leaky-relu edge logits (atomic u32 max on ordered encoding).
// ---------------------------------------------------------------------------
__global__ void gat_edge_max(const int* __restrict__ ei, const float* __restrict__ asrc,
                             const float* __restrict__ adst, unsigned* __restrict__ menc) {
  const int e = blockIdx.x * blockDim.x + threadIdx.x;
  if (e >= ETOT) return;
  int s, d; edge_sd(ei, e, s, d);
  const float e0 = lrelu(asrc[s * 2]     + adst[d * 2]);
  const float e1 = lrelu(asrc[s * 2 + 1] + adst[d * 2 + 1]);
  atomicMax(&menc[d * 2],     f2ord(e0));
  atomicMax(&menc[d * 2 + 1], f2ord(e1));
}

// ---------------------------------------------------------------------------
// ex = exp(e - max[dst]); stash per-edge, atomic-add into denom[dst].
// ---------------------------------------------------------------------------
__global__ void gat_edge_expsum(const int* __restrict__ ei, const float* __restrict__ asrc,
                                const float* __restrict__ adst, const unsigned* __restrict__ menc,
                                float* __restrict__ denom, float* __restrict__ exbuf) {
  const int e = blockIdx.x * blockDim.x + threadIdx.x;
  if (e >= ETOT) return;
  int s, d; edge_sd(ei, e, s, d);
  const float e0 = lrelu(asrc[s * 2]     + adst[d * 2]);
  const float e1 = lrelu(asrc[s * 2 + 1] + adst[d * 2 + 1]);
  const float x0 = __expf(e0 - ord2f(menc[d * 2]));
  const float x1 = __expf(e1 - ord2f(menc[d * 2 + 1]));
  exbuf[e * 2] = x0; exbuf[e * 2 + 1] = x1;
  atomicAdd(&denom[d * 2],     x0);
  atomicAdd(&denom[d * 2 + 1], x1);
}

// ---------------------------------------------------------------------------
// Weighted scatter: agg[dst] += s[src] * alpha. Wave32 per edge, each lane
// covers channels {l, l+32, l+64, l+96} (heads 0,0,1,1). L2-resident atomics.
// ---------------------------------------------------------------------------
__global__ __launch_bounds__(256) void gat_edge_agg(const int* __restrict__ ei,
                                                    const float* __restrict__ S,
                                                    const float* __restrict__ exbuf,
                                                    const float* __restrict__ denom,
                                                    float* __restrict__ agg) {
  const int e = blockIdx.x * 8 + (threadIdx.x >> 5);
  if (e >= ETOT) return;
  const int l = threadIdx.x & 31;
  int s, d; edge_sd(ei, e, s, d);
  const float a0 = exbuf[e * 2]     / denom[d * 2];
  const float a1 = exbuf[e * 2 + 1] / denom[d * 2 + 1];
  const float* srow = S + s * FDIM;
  float* drow = agg + d * FDIM;
  atomicAdd(&drow[l],      srow[l]      * a0);
  atomicAdd(&drow[l + 32], srow[l + 32] * a0);
  atomicAdd(&drow[l + 64], srow[l + 64] * a1);
  atomicAdd(&drow[l + 96], srow[l + 96] * a1);
}

// act==0 -> elu (layer 0); act==1 -> relu (layer 1). Concat heads (identity
// layout). Writes f16 (only consumer is the next layer's WMMA GEMM).
__global__ void gat_epilogue_concat(const float* __restrict__ agg, const float* __restrict__ bias,
                                    _Float16* __restrict__ Hf, int act) {
  const int i = blockIdx.x * blockDim.x + threadIdx.x;
  if (i >= NNODES * FDIM) return;
  const float v = agg[i] + bias[i & (FDIM - 1)];
  Hf[i] = (_Float16)(act ? fmaxf(v, 0.f) : (v > 0.f ? v : __expf(v) - 1.f));
}

// Final layer: mean over 2 heads, + bias, relu. f32 output.
__global__ void gat_epilogue_mean(const float* __restrict__ agg, const float* __restrict__ bias,
                                  float* __restrict__ out) {
  const int i = blockIdx.x * blockDim.x + threadIdx.x;
  if (i >= NNODES * 64) return;
  const int n = i >> 6, c = i & 63;
  const float v = 0.5f * (agg[n * FDIM + c] + agg[n * FDIM + 64 + c]) + bias[c];
  out[i] = fmaxf(v, 0.f);
}

extern "C" void kernel_launch(void* const* d_in, const int* in_sizes, int n_in,
                              void* d_out, int out_size, void* d_ws, size_t ws_size,
                              hipStream_t stream) {
  (void)in_sizes; (void)n_in; (void)out_size; (void)ws_size;
  const float* X   = (const float*)d_in[0];
  const int*   ei  = (const int*)  d_in[1];
  const float* W[3]  = { (const float*)d_in[2], (const float*)d_in[6],  (const float*)d_in[10] };
  const float* AS[3] = { (const float*)d_in[3], (const float*)d_in[7],  (const float*)d_in[11] };
  const float* AD[3] = { (const float*)d_in[4], (const float*)d_in[8],  (const float*)d_in[12] };
  const float* B[3]  = { (const float*)d_in[5], (const float*)d_in[9],  (const float*)d_in[13] };
  float* out = (float*)d_out;

  // Workspace carve-up (~92 MB, L2-resident on MI455X's 192 MB L2).
  // Wswz first so its 32B fragment loads stay aligned.
  char* p = (char*)d_ws;
  _Float16* Wswz = (_Float16*)p;            p += 32 * 32 * 16 * sizeof(_Float16); // 32 KB
  float*    S     = (float*)p;              p += (size_t)NNODES * FDIM * sizeof(float);
  float*    AGG   = (float*)p;              p += (size_t)NNODES * FDIM * sizeof(float);
  float*    asrc  = (float*)p;              p += (size_t)NNODES * 2 * sizeof(float);
  float*    adst  = (float*)p;              p += (size_t)NNODES * 2 * sizeof(float);
  unsigned* menc  = (unsigned*)p;           p += (size_t)NNODES * 2 * sizeof(unsigned);
  float*    denom = (float*)p;              p += (size_t)NNODES * 2 * sizeof(float);
  float*    exbuf = (float*)p;              p += (size_t)ETOT * 2 * sizeof(float);
  _Float16* Xf    = (_Float16*)p;           p += (size_t)NNODES * FDIM * sizeof(_Float16);
  _Float16* Hf    = (_Float16*)p;

  const dim3 gemm_grid(NNODES / 16), gemm_blk(64);
  const int init_blks  = (NNODES * FDIM + 255) / 256;
  const int alpha_blks = (NNODES + 7) / 8;
  const int edge_blks  = (ETOT + 255) / 256;
  const int eagg_blks  = (ETOT + 7) / 8;

  gat_cvt_f16<<<init_blks, 256, 0, stream>>>(X, Xf, NNODES * FDIM);

  auto run_layer = [&](const _Float16* in, int li) {
    gat_swizzle_w<<<(32 * 32 * 16 + 255) / 256, 256, 0, stream>>>(W[li], Wswz);
    gat_gemm_wmma<<<gemm_grid, gemm_blk, 0, stream>>>(in, Wswz, S);
    gat_init<<<init_blks, 256, 0, stream>>>(AGG, menc, denom);
    gat_alpha<<<alpha_blks, 256, 0, stream>>>(S, AS[li], AD[li], asrc, adst);
    gat_edge_max<<<edge_blks, 256, 0, stream>>>(ei, asrc, adst, menc);
    gat_edge_expsum<<<edge_blks, 256, 0, stream>>>(ei, asrc, adst, menc, denom, exbuf);
    gat_edge_agg<<<eagg_blks, 256, 0, stream>>>(ei, S, exbuf, denom, AGG);
  };

  run_layer(Xf, 0);
  gat_epilogue_concat<<<init_blks, 256, 0, stream>>>(AGG, B[0], Hf, /*act=elu*/0);
  run_layer(Hf, 1);
  gat_epilogue_concat<<<init_blks, 256, 0, stream>>>(AGG, B[1], Hf, /*act=relu*/1);
  run_layer(Hf, 2);
  gat_epilogue_mean<<<(NNODES * 64 + 255) / 256, 256, 0, stream>>>(AGG, B[2], out);
}
